// MiddleLayer_83580063580715
// MI455X (gfx1250) — matrix operand order repaired and checked by reading, not compiled, not run on hardware
//
#include <hip/hip_runtime.h>

typedef __attribute__((ext_vector_type(16))) __bf16 v16bf;
typedef __attribute__((ext_vector_type(8)))  float  v8f;
typedef __attribute__((ext_vector_type(2)))  __bf16 bf2;
typedef int vi4 __attribute__((vector_size(16)));   // matches builtin's b128 param

#define GLB_I32  __attribute__((address_space(1))) int*
#define LDS_I32  __attribute__((address_space(3))) int*
#define GLB_I128 __attribute__((address_space(1))) vi4*
#define LDS_I128 __attribute__((address_space(3))) vi4*

#if __has_builtin(__builtin_amdgcn_global_load_async_to_lds_b32) && \
    __has_builtin(__builtin_amdgcn_global_load_async_to_lds_b128)
#define USE_ASYNC_LDS 1
#else
#define USE_ASYNC_LDS 0
#endif

namespace {
constexpr int kBatch = 4096;   // M
constexpr int kNHalf = 2048;   // DE_MID / 2
constexpr int kK     = 2048;   // reduction length (EN_MID / 2)
constexpr int kLdIn  = 4096;   // in_data row stride (floats)
constexpr int kLdW   = 2048;   // W row stride (floats)
constexpr int kLdOut = 4096;   // out row stride (floats)

constexpr int TILE_M = 128;
constexpr int TILE_N = 128;    // per half (left/right)
constexpr int KSTEP  = 32;
constexpr int NIT    = kK / KSTEP;   // 64
constexpr int ROWF   = 36;     // padded floats per LDS row -> conflict-free b128

union FragBF { uint4 u[2]; v16bf v; };

__device__ inline unsigned cvt2bf(float lo, float hi) {
  bf2 t; t.x = (__bf16)lo; t.y = (__bf16)hi;   // hardware f32->bf16 (RNE)
  unsigned r; __builtin_memcpy(&r, &t, 4);
  return r;
}
} // namespace

__global__ __launch_bounds__(256) void
middlelayer_fused_bf16_wmma(const float* __restrict__ in_data,
                            const float* __restrict__ W,
                            const float* __restrict__ bias,
                            float* __restrict__ out)
{
  // double-buffered f32 staging tiles
  __shared__ __align__(16) float As[2][TILE_M * ROWF];       // 2 x 18 KB
  __shared__ __align__(16) float Bs[2][2 * TILE_N * ROWF];   // 2 x 36 KB

  const int tid  = threadIdx.x;
  const int lane = tid & 31;
  const int half = lane >> 4;
  const int l16  = lane & 15;
  const int wave = tid >> 5;

  const int tileM = blockIdx.y * TILE_M;
  const int cn    = blockIdx.x * TILE_N;

  // 8 waves -> 2 (M) x 4 (N); wave tile = 64M x 32N per half
  const int wm0 = (wave >> 2) * 64;
  const int wn0 = (wave & 3) * 32;

  // ---- staging assignments ----
  // A: lane gathers odd columns; element (m, k), m = tid/32 + 8*j, k = tid%32
  const int amBase = tid >> 5;              // 0..7
  const int ak     = tid & 31;
  const float* aSrcBase = in_data + (size_t)(tileM + amBase) * kLdIn + 2 * ak + 1;
  const int aLdsBase = amBase * ROWF + ak;  // float index

  // B: thread stages one 32-float k-slice of one W row (both halves covered)
  const int brow = (tid >> 7) ? (kNHalf + cn + (tid & 127)) : (cn + (tid & 127));
  const float* bSrc = W + (size_t)brow * kLdW;
  const int bLdsBase = tid * ROWF;

  auto stage = [&](int buf, int k0) {
#if USE_ASYNC_LDS
    const float* g = aSrcBase + 2 * k0;
    float* l = &As[buf][aLdsBase];
#pragma unroll
    for (int j = 0; j < 16; ++j) {
      __builtin_amdgcn_global_load_async_to_lds_b32(
          (GLB_I32)g, (LDS_I32)l, 0, 0);
      g += 8 * kLdIn;
      l += 8 * ROWF;
    }
    const float* gb = bSrc + k0;
    float* lb = &Bs[buf][bLdsBase];
#pragma unroll
    for (int i = 0; i < 8; ++i) {
      __builtin_amdgcn_global_load_async_to_lds_b128(
          (GLB_I128)(gb + 4 * i), (LDS_I128)(lb + 4 * i), 0, 0);
    }
#else
    const float* g = aSrcBase + 2 * k0;
    float av[16];
#pragma unroll
    for (int j = 0; j < 16; ++j) av[j] = g[(size_t)j * 8 * kLdIn];
#pragma unroll
    for (int j = 0; j < 16; ++j) As[buf][aLdsBase + j * 8 * ROWF] = av[j];
    const float4* gb = (const float4*)(bSrc + k0);
    float4 q[8];
#pragma unroll
    for (int i = 0; i < 8; ++i) q[i] = gb[i];
#pragma unroll
    for (int i = 0; i < 8; ++i) *(float4*)&Bs[buf][bLdsBase + 4 * i] = q[i];
#endif
  };

  v8f accL[4][2], accR[4][2];
#pragma unroll
  for (int mi = 0; mi < 4; ++mi)
#pragma unroll
    for (int nj = 0; nj < 2; ++nj) {
      accL[mi][nj] = (v8f){0.f,0.f,0.f,0.f,0.f,0.f,0.f,0.f};
      accR[mi][nj] = (v8f){0.f,0.f,0.f,0.f,0.f,0.f,0.f,0.f};
    }

  stage(0, 0);

  for (int it = 0; it < NIT; ++it) {
    const int buf = it & 1;
    __syncthreads();   // all consumers of buf^1 (iter it-1) finished reading
    if (it + 1 < NIT) stage(buf ^ 1, (it + 1) * KSTEP);   // async fill of buf^1
#if USE_ASYNC_LDS
    if (it + 1 < NIT) {
      // allow the 24 just-issued loads of the NEXT stage to remain in flight;
      // in-order completion => all of stage(it) has landed in LDS
      asm volatile("s_wait_asynccnt 0x18" ::: "memory");
    } else {
      asm volatile("s_wait_asynccnt 0x0" ::: "memory");
    }
#endif
    __syncthreads();   // buf fully staged and visible to all waves

    // ---- build bf16 fragments from f32 LDS tiles (ISA 7.12.2 layouts) ----
    FragBF a[4], bL[2], bR[2];
#pragma unroll
    for (int mi = 0; mi < 4; ++mi) {
      const float4* p =
          (const float4*)&As[buf][(wm0 + 16 * mi + l16) * ROWF + 8 * half];
      float4 f0 = p[0], f1 = p[1], f2 = p[4], f3 = p[5];   // K0.. and 16+K0..
      a[mi].u[0] = make_uint4(cvt2bf(f0.x, f0.y), cvt2bf(f0.z, f0.w),
                              cvt2bf(f1.x, f1.y), cvt2bf(f1.z, f1.w));
      a[mi].u[1] = make_uint4(cvt2bf(f2.x, f2.y), cvt2bf(f2.z, f2.w),
                              cvt2bf(f3.x, f3.y), cvt2bf(f3.z, f3.w));
    }
#pragma unroll
    for (int nj = 0; nj < 2; ++nj) {
      const float4* pl =
          (const float4*)&Bs[buf][(wn0 + 16 * nj + l16) * ROWF + 8 * half];
      float4 f0 = pl[0], f1 = pl[1], f2 = pl[4], f3 = pl[5];
      bL[nj].u[0] = make_uint4(cvt2bf(f0.x, f0.y), cvt2bf(f0.z, f0.w),
                               cvt2bf(f1.x, f1.y), cvt2bf(f1.z, f1.w));
      bL[nj].u[1] = make_uint4(cvt2bf(f2.x, f2.y), cvt2bf(f2.z, f2.w),
                               cvt2bf(f3.x, f3.y), cvt2bf(f3.z, f3.w));
      const float4* pr = (const float4*)
          &Bs[buf][(TILE_N + wn0 + 16 * nj + l16) * ROWF + 8 * half];
      float4 g0 = pr[0], g1 = pr[1], g2 = pr[4], g3 = pr[5];
      bR[nj].u[0] = make_uint4(cvt2bf(g0.x, g0.y), cvt2bf(g0.z, g0.w),
                               cvt2bf(g1.x, g1.y), cvt2bf(g1.z, g1.w));
      bR[nj].u[1] = make_uint4(cvt2bf(g2.x, g2.y), cvt2bf(g2.z, g2.w),
                               cvt2bf(g3.x, g3.y), cvt2bf(g3.z, g3.w));
    }

    // ---- 16 WMMA per K-step ----
#pragma unroll
    for (int mi = 0; mi < 4; ++mi)
#pragma unroll
      for (int nj = 0; nj < 2; ++nj) {
        accL[mi][nj] = __builtin_amdgcn_wmma_f32_16x16x32_bf16(
            false, a[mi].v, false, bL[nj].v, (short)0, accL[mi][nj], false, false);
        accR[mi][nj] = __builtin_amdgcn_wmma_f32_16x16x32_bf16(
            false, a[mi].v, false, bR[nj].v, (short)0, accR[mi][nj], false, false);
      }
  }

  // ---- epilogue: bias + relu, left - right, interleave with zero imag ----
#pragma unroll
  for (int nj = 0; nj < 2; ++nj) {
    const int c = cn + wn0 + 16 * nj + l16;     // output pair index
    const float biasL = bias[c];
    const float biasR = bias[c + kNHalf];
#pragma unroll
    for (int mi = 0; mi < 4; ++mi) {
      const int mbase = tileM + wm0 + 16 * mi + 8 * half;   // C-layout: M = r + 8*half
      float* po = out + (size_t)mbase * kLdOut + 2 * c;
#pragma unroll
      for (int r = 0; r < 8; ++r) {
        float vl = accL[mi][nj][r] + biasL; vl = vl > 0.f ? vl : 0.f;
        float vr = accR[mi][nj][r] + biasR; vr = vr > 0.f ? vr : 0.f;
        float2 res; res.x = vl - vr; res.y = 0.f;
        *(float2*)(po + (size_t)r * kLdOut) = res;
      }
    }
  }
}

extern "C" void kernel_launch(void* const* d_in, const int* in_sizes, int n_in,
                              void* d_out, int out_size, void* d_ws, size_t ws_size,
                              hipStream_t stream) {
  const float* in_data = (const float*)d_in[0];
  const float* W       = (const float*)d_in[1];
  const float* bias    = (const float*)d_in[2];
  float* out           = (float*)d_out;
  (void)in_sizes; (void)n_in; (void)out_size; (void)d_ws; (void)ws_size;

  dim3 grid(kNHalf / TILE_N, kBatch / TILE_M);   // (16, 32)
  middlelayer_fused_bf16_wmma<<<grid, 256, 0, stream>>>(in_data, W, bias, out);
}
